// FusionModule_64355789963982
// MI455X (gfx1250) — compile-verified, matrix operand order checked by reading, not run on hardware
//
#include <hip/hip_runtime.h>

// ---------------------------------------------------------------------------
// Problem constants (from reference)
// ---------------------------------------------------------------------------
#define BATCH 16
#define CIN   512
#define NTOK  1600      // 40*40
#define HC    64
#define NCHUNK (NTOK / 32)
#define SCALE 0.125f    // HC^-0.5, folded into Q

// Padded LDS row strides (halves) so ds_load_b128 fragment reads are
// conflict-free: K rows 64+8=72 (144B), V rows 32+8=40 (80B); both 16B aligned.
#define KROW 72
#define VROW 40

typedef _Float16 v16h __attribute__((ext_vector_type(16)));
typedef _Float16 v8h  __attribute__((ext_vector_type(8)));
typedef float    v8f  __attribute__((ext_vector_type(8)));
typedef unsigned int u32x4 __attribute__((ext_vector_type(4)));
typedef int          i32x4 __attribute__((ext_vector_type(4)));
typedef int          i32x8 __attribute__((ext_vector_type(8)));

#if __has_builtin(__builtin_amdgcn_tensor_load_to_lds) && __has_builtin(__builtin_amdgcn_s_wait_tensorcnt)
#define USE_TDM 1
#else
#define USE_TDM 0
#endif

// Load a 16-lane-striped WMMA fragment (A or B role) from a row-major f16 row.
// CDNA5 16-bit striping: lanes 0-15 hold K = {0..7, 16..23},
// lanes 16-31 hold K = {8..15, 24..31}  -> two contiguous 16B chunks per lane.
__device__ __forceinline__ v16h load_frag(const _Float16* row, int khalf) {
    v8h lo = *(const v8h*)(row + khalf * 8);
    v8h hi = *(const v8h*)(row + 16 + khalf * 8);
    return __builtin_shufflevector(lo, hi, 0,1,2,3,4,5,6,7,8,9,10,11,12,13,14,15);
}

// ---- VALU-only 16-lane reductions via DPP (no LDS round trips) ------------
// quad_perm(1,0,3,2)=0xB1 (xor1), quad_perm(2,3,0,1)=0x4E (xor2),
// row_half_mirror=0x141 (xor7 == xor4 once quads are merged),
// row_mirror=0x140 (xor15 == xor8 once 8-groups are merged).
template <int CTRL>
__device__ __forceinline__ float dpp_f(float v) {
    return __int_as_float(
        __builtin_amdgcn_update_dpp(0, __float_as_int(v), CTRL, 0xf, 0xf, true));
}
__device__ __forceinline__ float red_max16(float v) {
    v = fmaxf(v, dpp_f<0xB1>(v));
    v = fmaxf(v, dpp_f<0x4E>(v));
    v = fmaxf(v, dpp_f<0x141>(v));
    v = fmaxf(v, dpp_f<0x140>(v));
    return v;
}
__device__ __forceinline__ float red_sum16(float v) {
    v += dpp_f<0xB1>(v);
    v += dpp_f<0x4E>(v);
    v += dpp_f<0x141>(v);
    v += dpp_f<0x140>(v);
    return v;
}

// ---------------------------------------------------------------------------
// Kernel 1: QKV projection.
//   Q[b,n,d] = 0.125 * sum_c rgb[b,c,n] * wq[c,d]   -> f16, row-major [B][N][HC]
//   K[b,n,d] =          sum_c rgb[b,c,n] * wk[c,d]  -> f16, row-major [B][N][HC]
//   Vt[b,d,n] =         sum_c hsi[b,c,n] * wv[c,d]  -> f16, [B][HC][N] (transposed)
// ---------------------------------------------------------------------------
__global__ __launch_bounds__(256)
void proj_qkv_kernel(const float* __restrict__ rgb, const float* __restrict__ hsi,
                     const float* __restrict__ wq,  const float* __restrict__ wk,
                     const float* __restrict__ wv,
                     _Float16* __restrict__ Qh, _Float16* __restrict__ Kh,
                     _Float16* __restrict__ Vth) {
    const int b     = blockIdx.x / (NTOK / 64);
    const int ntile = (blockIdx.x % (NTOK / 64)) * 64;
    const int tid   = threadIdx.x;
    const int t     = tid & 63;       // token within tile
    const int dq    = tid >> 6;       // which 16-wide slice of HC

    __shared__ float s_wq[64 * 64];
    __shared__ float s_wk[64 * 64];
    __shared__ float s_wv[64 * 64];

    float aq[16], ak[16], av[16];
#pragma unroll
    for (int j = 0; j < 16; ++j) { aq[j] = 0.f; ak[j] = 0.f; av[j] = 0.f; }

    for (int cb = 0; cb < CIN / 64; ++cb) {
        const int c0 = cb * 64;
        for (int i = tid; i < 64 * 64; i += 256) {
            const int c = c0 + (i >> 6), d = i & 63;
            s_wq[i] = wq[c * HC + d];
            s_wk[i] = wk[c * HC + d];
            s_wv[i] = wv[c * HC + d];
        }
        __syncthreads();

        for (int cl = 0; cl < 64; ++cl) {
            const size_t xoff = ((size_t)b * CIN + (c0 + cl)) * NTOK + ntile + t;
            const float xr = rgb[xoff];
            const float xh = hsi[xoff];
            const float* wqc = s_wq + cl * 64 + (dq << 4);
            const float* wkc = s_wk + cl * 64 + (dq << 4);
            const float* wvc = s_wv + cl * 64 + (dq << 4);
#pragma unroll
            for (int jj = 0; jj < 4; ++jj) {
                const float4 q4 = ((const float4*)wqc)[jj];
                const float4 k4 = ((const float4*)wkc)[jj];
                const float4 v4 = ((const float4*)wvc)[jj];
                aq[4*jj+0] += xr*q4.x; aq[4*jj+1] += xr*q4.y; aq[4*jj+2] += xr*q4.z; aq[4*jj+3] += xr*q4.w;
                ak[4*jj+0] += xr*k4.x; ak[4*jj+1] += xr*k4.y; ak[4*jj+2] += xr*k4.z; ak[4*jj+3] += xr*k4.w;
                av[4*jj+0] += xh*v4.x; av[4*jj+1] += xh*v4.y; av[4*jj+2] += xh*v4.z; av[4*jj+3] += xh*v4.w;
            }
        }
        __syncthreads();
    }

    const size_t rowbase = ((size_t)b * NTOK + ntile + t) * HC + (dq << 4);
    v8h h0, h1, g0, g1;
#pragma unroll
    for (int j = 0; j < 8; ++j) {
        h0[j] = (_Float16)(aq[j] * SCALE); h1[j] = (_Float16)(aq[j + 8] * SCALE);
        g0[j] = (_Float16)ak[j];           g1[j] = (_Float16)ak[j + 8];
    }
    *(v8h*)(Qh + rowbase)     = h0;  *(v8h*)(Qh + rowbase + 8) = h1;
    *(v8h*)(Kh + rowbase)     = g0;  *(v8h*)(Kh + rowbase + 8) = g1;
#pragma unroll
    for (int j = 0; j < 16; ++j)
        Vth[((size_t)b * HC + (dq << 4) + j) * NTOK + ntile + t] = (_Float16)av[j];
}

// ---------------------------------------------------------------------------
// Kernel 2: flash attention with v_wmma_f32_16x16x32_f16.
// Block = 4 waves, TWO 16-query tiles per wave (K/V fragments reused across
// both A matrices -> 16 WMMAs per chunk per wave). All tiles in one batch.
// K/V 32-key chunks double-buffered in LDS via the Tensor Data Mover with
// descriptor padding for conflict-free LDS row strides.
// Grid: (13, BATCH); tail tiles clamp to tile 99 (redundant identical work).
// ---------------------------------------------------------------------------
__global__ __launch_bounds__(128)
void flash_attn_kernel(const _Float16* __restrict__ Qh, const _Float16* __restrict__ Kh,
                       const _Float16* __restrict__ Vth, float* __restrict__ res) {
    const int lane  = threadIdx.x & 31;
    const int wid   = threadIdx.x >> 5;
    const int b     = blockIdx.y;
    int t0 = blockIdx.x * 8 + wid * 2;               // tile index within batch
    int t1 = t0 + 1;
    if (t0 > 99) t0 = 99;
    if (t1 > 99) t1 = 99;
    const int qtile0 = t0 * 16;
    const int qtile1 = t1 * 16;

    const int col   = lane & 15;
    const int khalf = lane >> 4;

    __shared__ _Float16 kbuf[2][32 * KROW];          // 32 keys x 64 (+8 pad)
    __shared__ _Float16 vbuf[2][HC * VROW];          // 64 hc   x 32 (+8 pad)
    __shared__ _Float16 lds_p[8][16 * 32];
    _Float16* pbuf0 = &lds_p[wid * 2][0];
    _Float16* pbuf1 = &lds_p[wid * 2 + 1][0];

#if USE_TDM
    // --- Tensor DMA descriptors: one K tile + one V tile per chunk ----------
    // K: 2D tile 64(elem) x 32(keys), stride 64, pad 4 DW per 32 DW  -> KROW
    // V: 2D tile 32(elem) x 64(hc),   stride 1600, pad 4 DW per 16 DW -> VROW
    const int kg1w0 = (1 << 16) | (1 << 20) | (4 << 22) | (3 << 25);
    const int vg1w0 = (1 << 16) | (1 << 20) | (3 << 22) | (3 << 25);
    const i32x8 g1k = { kg1w0, 64 << 16, 32 << 16, 64 << 16, 32, 64, 0, 0 };
    const i32x8 g1v = { vg1w0, 32 << 16, 64 << 16, 32 << 16, 64, 1600, 0, 0 };
    const i32x4 gz4 = { 0, 0, 0, 0 };
    const i32x8 gz8 = { 0, 0, 0, 0, 0, 0, 0, 0 };
#endif

    // Q fragments for both tiles (reused across all key chunks)
    const _Float16* qrow0 = Qh + ((size_t)b * NTOK + qtile0 + col) * HC;
    const _Float16* qrow1 = Qh + ((size_t)b * NTOK + qtile1 + col) * HC;
    const v16h aq00 = load_frag(qrow0, khalf);
    const v16h aq01 = load_frag(qrow0 + 32, khalf);
    const v16h aq10 = load_frag(qrow1, khalf);
    const v16h aq11 = load_frag(qrow1 + 32, khalf);

    auto stage = [&](int kb, int buf) {
        const int k0 = kb * 32;
#if USE_TDM
        if (wid == 0) {
            const unsigned long long gK =
                (unsigned long long)(size_t)(Kh + ((size_t)b * NTOK + k0) * HC);
            const unsigned long long gV =
                (unsigned long long)(size_t)(Vth + ((size_t)b * HC) * NTOK + k0);
            const unsigned ldsK = (unsigned)(size_t)&kbuf[buf][0];
            const unsigned ldsV = (unsigned)(size_t)&vbuf[buf][0];
            const u32x4 g0k = { 1u, ldsK, (unsigned)gK,
                                (unsigned)((gK >> 32) & 0x01FFFFFFu) | 0x80000000u };
            const u32x4 g0v = { 1u, ldsV, (unsigned)gV,
                                (unsigned)((gV >> 32) & 0x01FFFFFFu) | 0x80000000u };
            __builtin_amdgcn_tensor_load_to_lds(g0k, g1k, gz4, gz4, gz8, 0);
            __builtin_amdgcn_tensor_load_to_lds(g0v, g1v, gz4, gz4, gz8, 0);
        }
#else
        const int tid = threadIdx.x;
        for (int ch = tid; ch < 256; ch += 128) {          // K: 256 x 16B
            const int row = ch >> 2, part = ch & 3;
            *(v8h*)(&kbuf[buf][row * KROW + part * 8]) =
                *(const v8h*)(Kh + ((size_t)b * NTOK + k0 + row) * HC + part * 8);
        }
        for (int ch = tid; ch < 256; ch += 128) {          // V: 256 x 16B
            const int hc = ch >> 1, part = ch & 1;
            *(v8h*)(&vbuf[buf][hc * VROW + part * 8]) =
                *(const v8h*)(Vth + ((size_t)b * HC + hc) * NTOK + k0 + part * 8);
        }
#endif
    };

    v8f o00 = {}, o01 = {}, o02 = {}, o03 = {};
    v8f o10 = {}, o11 = {}, o12 = {}, o13 = {};
    float m0[8], l0[8], m1[8], l1[8];
#pragma unroll
    for (int r = 0; r < 8; ++r) {
        m0[r] = -1e30f; l0[r] = 0.f;
        m1[r] = -1e30f; l1[r] = 0.f;
    }

    stage(0, 0);                                     // prologue: chunk 0 -> buf0

    for (int kb = 0; kb < NCHUNK; ++kb) {
        const int cur = kb & 1;
        __syncthreads();                             // buf[cur^1] free to overwrite
        if (kb + 1 < NCHUNK) {
            stage(kb + 1, cur ^ 1);
#if USE_TDM
            if (wid == 0) __builtin_amdgcn_s_wait_tensorcnt(2);  // chunk kb landed
#endif
        }
#if USE_TDM
        else if (wid == 0) __builtin_amdgcn_s_wait_tensorcnt(0);
#endif
        __syncthreads();                             // buf[cur] visible to all waves

        // ----- K fragments loaded once, reused by both query tiles -----
        const _Float16* krow0 = &kbuf[cur][col * KROW];
        const _Float16* krow1 = &kbuf[cur][(col + 16) * KROW];
        const v16h bk00 = load_frag(krow0,      khalf);
        const v16h bk01 = load_frag(krow0 + 32, khalf);
        const v16h bk10 = load_frag(krow1,      khalf);
        const v16h bk11 = load_frag(krow1 + 32, khalf);

        v8f s00 = {}, s01 = {}, s10 = {}, s11 = {};
        s00 = __builtin_amdgcn_wmma_f32_16x16x32_f16(false, aq00, false, bk00, (short)0, s00, false, false);
        s00 = __builtin_amdgcn_wmma_f32_16x16x32_f16(false, aq01, false, bk01, (short)0, s00, false, false);
        s01 = __builtin_amdgcn_wmma_f32_16x16x32_f16(false, aq00, false, bk10, (short)0, s01, false, false);
        s01 = __builtin_amdgcn_wmma_f32_16x16x32_f16(false, aq01, false, bk11, (short)0, s01, false, false);
        s10 = __builtin_amdgcn_wmma_f32_16x16x32_f16(false, aq10, false, bk00, (short)0, s10, false, false);
        s10 = __builtin_amdgcn_wmma_f32_16x16x32_f16(false, aq11, false, bk01, (short)0, s10, false, false);
        s11 = __builtin_amdgcn_wmma_f32_16x16x32_f16(false, aq10, false, bk10, (short)0, s11, false, false);
        s11 = __builtin_amdgcn_wmma_f32_16x16x32_f16(false, aq11, false, bk11, (short)0, s11, false, false);

        // ----- online softmax for both tiles (independent chains interleave) ---
#pragma unroll
        for (int r = 0; r < 8; ++r) {
            const float mxa = red_max16(fmaxf(s00[r], s01[r]));
            const float mxb = red_max16(fmaxf(s10[r], s11[r]));
            const float mna = fmaxf(m0[r], mxa);
            const float mnb = fmaxf(m1[r], mxb);
            const float ala = __expf(m0[r] - mna);
            const float alb = __expf(m1[r] - mnb);
            m0[r] = mna; m1[r] = mnb;
            const float pa0 = __expf(s00[r] - mna);
            const float pa1 = __expf(s01[r] - mna);
            const float pb0 = __expf(s10[r] - mnb);
            const float pb1 = __expf(s11[r] - mnb);
            l0[r] = l0[r] * ala + red_sum16(pa0 + pa1);
            l1[r] = l1[r] * alb + red_sum16(pb0 + pb1);
            o00[r] *= ala; o01[r] *= ala; o02[r] *= ala; o03[r] *= ala;
            o10[r] *= alb; o11[r] *= alb; o12[r] *= alb; o13[r] *= alb;
            const int pr = (r + 8 * khalf) * 32 + col;
            pbuf0[pr]      = (_Float16)pa0;
            pbuf0[pr + 16] = (_Float16)pa1;
            pbuf1[pr]      = (_Float16)pb0;
            pbuf1[pr + 16] = (_Float16)pb1;
        }

        // ----- O += P @ V : V fragments loaded once, reused by both tiles -----
        const v16h ap0 = load_frag(pbuf0 + col * 32, khalf);
        const v16h ap1 = load_frag(pbuf1 + col * 32, khalf);
        const v16h bv0 = load_frag(&vbuf[cur][(col     ) * VROW], khalf);
        const v16h bv1 = load_frag(&vbuf[cur][(col + 16) * VROW], khalf);
        const v16h bv2 = load_frag(&vbuf[cur][(col + 32) * VROW], khalf);
        const v16h bv3 = load_frag(&vbuf[cur][(col + 48) * VROW], khalf);
        o00 = __builtin_amdgcn_wmma_f32_16x16x32_f16(false, ap0, false, bv0, (short)0, o00, false, false);
        o01 = __builtin_amdgcn_wmma_f32_16x16x32_f16(false, ap0, false, bv1, (short)0, o01, false, false);
        o02 = __builtin_amdgcn_wmma_f32_16x16x32_f16(false, ap0, false, bv2, (short)0, o02, false, false);
        o03 = __builtin_amdgcn_wmma_f32_16x16x32_f16(false, ap0, false, bv3, (short)0, o03, false, false);
        o10 = __builtin_amdgcn_wmma_f32_16x16x32_f16(false, ap1, false, bv0, (short)0, o10, false, false);
        o11 = __builtin_amdgcn_wmma_f32_16x16x32_f16(false, ap1, false, bv1, (short)0, o11, false, false);
        o12 = __builtin_amdgcn_wmma_f32_16x16x32_f16(false, ap1, false, bv2, (short)0, o12, false, false);
        o13 = __builtin_amdgcn_wmma_f32_16x16x32_f16(false, ap1, false, bv3, (short)0, o13, false, false);
    }

    // ----- finalize: divide by l, store res[b][row][hc] for both tiles -----
#pragma unroll
    for (int r = 0; r < 8; ++r) {
        const float ia = 1.0f / l0[r];
        const float ib = 1.0f / l1[r];
        const size_t ra = ((size_t)b * NTOK + qtile0 + r + 8 * khalf) * HC + col;
        const size_t rb = ((size_t)b * NTOK + qtile1 + r + 8 * khalf) * HC + col;
        res[ra]      = o00[r] * ia;
        res[ra + 16] = o01[r] * ia;
        res[ra + 32] = o02[r] * ia;
        res[ra + 48] = o03[r] * ia;
        res[rb]      = o10[r] * ib;
        res[rb + 16] = o11[r] * ib;
        res[rb + 32] = o12[r] * ib;
        res[rb + 48] = o13[r] * ib;
    }
}

// ---------------------------------------------------------------------------
// Kernel 3: output projection with the torch transpose/reshape reinterpretation:
//   res2[b,i,j] = res[b, (i*64+j) % 1600, (i*64+j) / 1600]
//   out[b,c,n]  = sum_j res2[b,n,j] * wo[j,c]
// ---------------------------------------------------------------------------
__global__ __launch_bounds__(256)
void out_proj_kernel(const float* __restrict__ res, const float* __restrict__ wo,
                     float* __restrict__ out) {
    const int b     = blockIdx.x / (NTOK / 64);
    const int ntile = (blockIdx.x % (NTOK / 64)) * 64;
    const int tid   = threadIdx.x;

    __shared__ float s_r[64 * 68];

    for (int i = tid; i < 64 * HC; i += 256) {
        const int tok = i >> 6, j = i & 63;
        const int f = (ntile + tok) * HC + j;
        const int m = f % NTOK, d = f / NTOK;
        s_r[tok * 68 + j] = res[((size_t)b * NTOK + m) * HC + d];
    }
    __syncthreads();

    const int n_local = tid & 63;
    const int group   = tid >> 6;
    const float* rrow = s_r + n_local * 68;

    for (int c = group * 128; c < group * 128 + 128; ++c) {
        float acc = 0.f;
#pragma unroll
        for (int jj = 0; jj < 16; ++jj) {
            const float4 rv = ((const float4*)rrow)[jj];
            acc += rv.x * wo[(4*jj+0) * CIN + c];
            acc += rv.y * wo[(4*jj+1) * CIN + c];
            acc += rv.z * wo[(4*jj+2) * CIN + c];
            acc += rv.w * wo[(4*jj+3) * CIN + c];
        }
        out[((size_t)b * CIN + c) * NTOK + ntile + n_local] = acc;
    }
}

// ---------------------------------------------------------------------------
// Launch: ws layout = Qh | Kh | Vth (f16) | res (f32)  ~ 15.6 MB
// ---------------------------------------------------------------------------
extern "C" void kernel_launch(void* const* d_in, const int* in_sizes, int n_in,
                              void* d_out, int out_size, void* d_ws, size_t ws_size,
                              hipStream_t stream) {
    const float* rgb = (const float*)d_in[0];
    const float* hsi = (const float*)d_in[1];
    const float* wq  = (const float*)d_in[2];
    const float* wk  = (const float*)d_in[3];
    const float* wv  = (const float*)d_in[4];
    const float* wo  = (const float*)d_in[5];
    float* out = (float*)d_out;

    const size_t nqkv = (size_t)BATCH * NTOK * HC;
    _Float16* Qh  = (_Float16*)d_ws;
    _Float16* Kh  = Qh + nqkv;
    _Float16* Vth = Kh + nqkv;
    float*    res = (float*)(Vth + nqkv);

    proj_qkv_kernel<<<BATCH * (NTOK / 64), 256, 0, stream>>>(rgb, hsi, wq, wk, wv, Qh, Kh, Vth);
    flash_attn_kernel<<<dim3(13, BATCH), 128, 0, stream>>>(Qh, Kh, Vth, res);
    out_proj_kernel<<<BATCH * (NTOK / 64), 256, 0, stream>>>(res, wo, out);
}
